// MLA_3985729651558
// MI455X (gfx1250) — matrix operand order repaired and checked by reading, not compiled, not run on hardware
//
#include <hip/hip_runtime.h>

// ---------------------------------------------------------------------------
// Types for CDNA5 WMMA (wave32, 16x16x32 bf16 -> f32)
// ---------------------------------------------------------------------------
typedef __bf16 bf16;
typedef __attribute__((ext_vector_type(16))) __bf16 v16bf;
typedef __attribute__((ext_vector_type(8)))  __bf16 v8bf;
typedef __attribute__((ext_vector_type(8)))  float  v8f;

// Problem constants (match reference)
constexpr int CB   = 2;
constexpr int CS   = 2048;
constexpr int CHID = 2048;
constexpr int CH   = 16;
constexpr int CQL  = 1536;
constexpr int CKVL = 512;
constexpr int CDR  = 64;
constexpr int CDN  = 128;
constexpr int CDV  = 128;
constexpr int CDQK = 192;            // 128 + 64
constexpr int CM   = CB * CS;        // 4096 token rows
constexpr int CKVP = 640;            // kv_a projection padded to N%128==0 (576 -> 640)
constexpr float CSCALE = 0.07216878364870323f; // 192^-0.5, folded into Q

__device__ __forceinline__ v8f wmma_bf16(v16bf a, v16bf b, v8f c) {
  // D = A(16x32) * B(32x16) + C ; emits v_wmma_f32_16x16x32_bf16
  return __builtin_amdgcn_wmma_f32_16x16x32_bf16(false, a, false, b,
                                                 (short)0, c, false, false);
}

// A-fragment (16x32, MxK): lane half h reads K = h*16 .. h*16+15
__device__ __forceinline__ v16bf load_a_frag(const bf16* A, int ldk, int mbase,
                                             int kbase, int lane) {
  const int m  = mbase + (lane & 15);
  const int hh = (lane >> 4) * 16;
  const bf16* p = A + (size_t)m * ldk + kbase + hh;
  v8bf lo = *(const v8bf*)p;
  v8bf hi = *(const v8bf*)(p + 8);
  return __builtin_shufflevector(lo, hi, 0,1,2,3,4,5,6,7,8,9,10,11,12,13,14,15);
}

// B-fragment (32x16, KxN): lane half h reads contraction K = h*16 .. h*16+15,
// contiguous along the source row (row index n already includes lane&15).
__device__ __forceinline__ v16bf load_b_frag(const bf16* Bm, int ldk, int n,
                                             int kbase, int lane) {
  const bf16* p = Bm + (size_t)n * ldk + kbase + ((lane >> 4) << 4);
  return *(const v16bf*)p;
}

// ---------------------------------------------------------------------------
// fp32 -> bf16 conversion; zero-pads [n, ntot) so padded GEMM operands are 0.
// ---------------------------------------------------------------------------
__global__ __launch_bounds__(256)
void cvt_bf16_pad(const float* __restrict__ in, bf16* __restrict__ out,
                  long n, long ntot) {
  long i = (long)blockIdx.x * blockDim.x + threadIdx.x;
  const long stride = (long)gridDim.x * blockDim.x;
  for (; i < ntot; i += stride) out[i] = (i < n) ? (bf16)in[i] : (bf16)0.f;
}

// ---------------------------------------------------------------------------
// Generic NT GEMM: C[M,N](f32) = A[M,K](bf16) * B[N,K](bf16)^T
// Requires M % 128 == 0, N % 128 == 0, K % 64 == 0 (call sites guarantee it;
// the one odd shape, N=576, is zero-padded to 640).
// Block = 256 thr = 8 waves; block tile 128x128; wave tile 32x64.
// Register double-buffered over K: the fragment loads for chunk k+32 are in
// flight while the 8 WMMAs for chunk k execute. No divergent guards in the
// hot loop, so the 12 b128 loads can clause and take partial loadcnt waits.
// ---------------------------------------------------------------------------
__device__ __forceinline__ void gemm_load_tiles(
    const bf16* __restrict__ A, const bf16* __restrict__ Bw, int K,
    int m0, int n0, int lane, int kb, v16bf af[2], v16bf bfr[4]) {
  #pragma unroll
  for (int t = 0; t < 2; ++t) af[t] = load_a_frag(A, K, m0 + t * 16, kb, lane);
  const int l15 = lane & 15;
  #pragma unroll
  for (int u = 0; u < 4; ++u)
    bfr[u] = load_b_frag(Bw, K, n0 + u * 16 + l15, kb, lane);
}

__global__ __launch_bounds__(256)
void gemm_nt_bf16(const bf16* __restrict__ A, const bf16* __restrict__ Bw,
                  float* __restrict__ C, int M, int N, int K) {
  (void)M;
  const int lane = threadIdx.x & 31;
  const int wave = threadIdx.x >> 5;
  const int m0 = blockIdx.y * 128 + (wave >> 1) * 32;
  const int n0 = blockIdx.x * 128 + (wave & 1) * 64;
  const int l15  = lane & 15;
  const int half = lane >> 4;

  v8f acc[2][4];
  #pragma unroll
  for (int t = 0; t < 2; ++t)
    #pragma unroll
    for (int u = 0; u < 4; ++u)
      #pragma unroll
      for (int j = 0; j < 8; ++j) acc[t][u][j] = 0.f;

  v16bf a0[2], b0[4], a1[2], b1[4];
  gemm_load_tiles(A, Bw, K, m0, n0, lane, 0, a0, b0);

  for (int kb = 0; kb < K; kb += 64) {
    // L2 prefetch two chunks ahead (uniform branch, no exec masking)
    if (kb + 128 < K) {
      __builtin_prefetch(A + (size_t)(m0 + l15) * K + kb + 128, 0, 1);
      __builtin_prefetch(Bw + (size_t)(n0 + l15) * K + kb + 128, 0, 1);
    }
    // stage 1: loads for kb+32 fly while WMMAs for kb execute
    gemm_load_tiles(A, Bw, K, m0, n0, lane, kb + 32, a1, b1);
    #pragma unroll
    for (int t = 0; t < 2; ++t)
      #pragma unroll
      for (int u = 0; u < 4; ++u)
        acc[t][u] = wmma_bf16(a0[t], b0[u], acc[t][u]);
    // stage 2: loads for kb+64 fly while WMMAs for kb+32 execute
    if (kb + 64 < K)
      gemm_load_tiles(A, Bw, K, m0, n0, lane, kb + 64, a0, b0);
    #pragma unroll
    for (int t = 0; t < 2; ++t)
      #pragma unroll
      for (int u = 0; u < 4; ++u)
        acc[t][u] = wmma_bf16(a1[t], b1[u], acc[t][u]);
  }

  // C-fragment layout: row = m0 + t*16 + 8*half + vgpr, col = n0 + u*16 + l15
  #pragma unroll
  for (int t = 0; t < 2; ++t)
    #pragma unroll
    for (int u = 0; u < 4; ++u) {
      const int col = n0 + u * 16 + l15;
      #pragma unroll
      for (int j = 0; j < 8; ++j) {
        const int row = m0 + t * 16 + half * 8 + j;
        C[(size_t)row * N + col] = acc[t][u][j];
      }
    }
}

// ---------------------------------------------------------------------------
// RMSNorm (f32 in, strided) + convert to bf16; one block per row.
// ---------------------------------------------------------------------------
__global__ __launch_bounds__(256)
void rmsnorm_cvt(const float* __restrict__ in, int ldin,
                 const float* __restrict__ w, bf16* __restrict__ out, int N) {
  const int row = blockIdx.x;
  const float* r = in + (size_t)row * ldin;
  float ss = 0.f;
  for (int i = threadIdx.x; i < N; i += blockDim.x) { float v = r[i]; ss += v * v; }
  #pragma unroll
  for (int m = 16; m >= 1; m >>= 1) ss += __shfl_xor(ss, m, 32);
  __shared__ float red[8];
  const int lane = threadIdx.x & 31, wv = threadIdx.x >> 5;
  if (lane == 0) red[wv] = ss;
  __syncthreads();
  float tot = 0.f;
  #pragma unroll
  for (int i = 0; i < 8; ++i) tot += red[i];
  const float inv = rsqrtf(tot / (float)N + 1e-6f);
  for (int i = threadIdx.x; i < N; i += blockDim.x)
    out[(size_t)row * N + i] = (bf16)(r[i] * inv * w[i]);
}

// ---------------------------------------------------------------------------
// RoPE on q_pe + pack q f32[BS, H*192] -> Qb bf16[(B*H), S, 192], scale folded
// ---------------------------------------------------------------------------
__global__ __launch_bounds__(128)
void rope_q_pack(const float* __restrict__ q, const float* __restrict__ fc,
                 bf16* __restrict__ Qb) {
  const int idx = blockIdx.x;           // (b*S + s)*H + h
  const int h  = idx % CH;
  const int bs = idx / CH;
  const int s  = bs % CS;
  const int b  = bs / CS;
  const float* src = q + (size_t)bs * (CH * CDQK) + h * CDQK;
  bf16* dst = Qb + ((size_t)(b * CH + h) * CS + s) * CDQK;
  const int t = threadIdx.x;
  if (t < CDN) dst[t] = (bf16)(src[t] * CSCALE);
  if (t < CDR / 2) {
    const float c  = fc[s * CDR + 2 * t];
    const float sn = fc[s * CDR + 2 * t + 1];
    const float xr = src[CDN + 2 * t];
    const float xi = src[CDN + 2 * t + 1];
    dst[CDN + 2 * t]     = (bf16)((xr * c - xi * sn) * CSCALE);
    dst[CDN + 2 * t + 1] = (bf16)((xr * sn + xi * c) * CSCALE);
  }
}

// ---------------------------------------------------------------------------
// Pack kv f32[BS, H*256] -> Kb bf16[(B*H), S, 192] (k_nope + broadcast rope'd
// k_pe from kv_all cols 512..575, padded stride) and Vt bf16[(B*H), 128, S].
// ---------------------------------------------------------------------------
__global__ __launch_bounds__(128)
void kv_pack(const float* __restrict__ kv, const float* __restrict__ kvall,
             const float* __restrict__ fc, bf16* __restrict__ Kb,
             bf16* __restrict__ Vt) {
  const int idx = blockIdx.x;           // (b*S + s)*H + h
  const int h  = idx % CH;
  const int bs = idx / CH;
  const int s  = bs % CS;
  const int b  = bs / CS;
  const float* src = kv + (size_t)bs * (CH * (CDN + CDV)) + h * (CDN + CDV);
  bf16* kdst = Kb + ((size_t)(b * CH + h) * CS + s) * CDQK;
  const int t = threadIdx.x;
  if (t < CDN) {
    kdst[t] = (bf16)src[t];
    Vt[((size_t)(b * CH + h) * CDV + t) * CS + s] = (bf16)src[CDN + t];
  }
  if (t < CDR / 2) {
    const float c  = fc[s * CDR + 2 * t];
    const float sn = fc[s * CDR + 2 * t + 1];
    const float* pe = kvall + (size_t)bs * CKVP + CKVL;   // padded row stride
    const float xr = pe[2 * t];
    const float xi = pe[2 * t + 1];
    kdst[CDN + 2 * t]     = (bf16)(xr * c - xi * sn);
    kdst[CDN + 2 * t + 1] = (bf16)(xr * sn + xi * c);
  }
}

// ---------------------------------------------------------------------------
// Causal flash attention. One wave per 16-query tile of one (b,h).
// Q: 6 A-frags (d=192); scores via WMMA; online softmax with 16-lane shfl
// reductions; P staged through per-wave LDS tile to re-enter A-frag layout;
// P*V via WMMA into 16x128 f32 accumulator. Next chunk K/V prefetched while
// the current chunk's softmax + LDS transpose executes.
// ---------------------------------------------------------------------------
constexpr int ATT_WAVES = 4;

__global__ __launch_bounds__(32 * ATT_WAVES)
void mla_attn(const bf16* __restrict__ Qb, const bf16* __restrict__ Kb,
              const bf16* __restrict__ Vt, bf16* __restrict__ Oc) {
  __shared__ bf16 Pld[ATT_WAVES][16 * 32];
  const int lane = threadIdx.x & 31;
  const int wv   = threadIdx.x >> 5;
  const int task = blockIdx.x * ATT_WAVES + wv;   // B*H*(S/16) tasks
  const int qt   = task & (CS / 16 - 1);
  const int bh   = task / (CS / 16);
  const int b    = bh / CH;
  const int h    = bh % CH;
  const int qbase = qt * 16;
  const int half = lane >> 4;
  const int l15  = lane & 15;

  const bf16* Qp = Qb + (size_t)bh * CS * CDQK;
  const bf16* Kp = Kb + (size_t)bh * CS * CDQK;
  const bf16* Vp = Vt + (size_t)bh * CDV * CS;
  bf16* P = &Pld[wv][0];

  v16bf qf[6];
  #pragma unroll
  for (int dc = 0; dc < 6; ++dc) qf[dc] = load_a_frag(Qp, CDQK, qbase, dc * 32, lane);

  float mrow[8], lrow[8];
  v8f o[8];
  #pragma unroll
  for (int j = 0; j < 8; ++j) { mrow[j] = -1e30f; lrow[j] = 0.f; }
  #pragma unroll
  for (int n = 0; n < 8; ++n)
    #pragma unroll
    for (int j = 0; j < 8; ++j) o[n][j] = 0.f;

  const int kmax = qbase + 16;                    // causal: keys <= last query
  for (int kb = 0; kb < kmax; kb += 32) {
    // prefetch next chunk's K rows and V columns while this chunk computes
    if (kb + 32 < kmax) {
      __builtin_prefetch(Kp + (size_t)(kb + 32 + lane) * CDQK, 0, 1);
      __builtin_prefetch(Vp + (size_t)(lane * 4) * CS + kb + 32, 0, 1);
    }

    float sreg[2][8];
    float tmax[8];
    #pragma unroll
    for (int j = 0; j < 8; ++j) tmax[j] = -1e30f;

    #pragma unroll
    for (int t = 0; t < 2; ++t) {
      const int kt = kb + t * 16;
      if (kt < kmax) {                            // wave-uniform branch
        v8f s;
        #pragma unroll
        for (int j = 0; j < 8; ++j) s[j] = 0.f;
        #pragma unroll
        for (int dc = 0; dc < 6; ++dc) {
          v16bf kf = load_b_frag(Kp, CDQK, kt + l15, dc * 32, lane);
          s = wmma_bf16(qf[dc], kf, s);
        }
        const int kj = kt + l15;
        #pragma unroll
        for (int j = 0; j < 8; ++j) {
          const int qi = qbase + half * 8 + j;
          const float sv = (kj <= qi) ? s[j] : -1e30f;
          sreg[t][j] = sv;
          tmax[j] = fmaxf(tmax[j], sv);
        }
      } else {
        #pragma unroll
        for (int j = 0; j < 8; ++j) sreg[t][j] = -1e30f;
      }
    }

    // row max across the 16 lanes holding this row
    #pragma unroll
    for (int msk = 8; msk >= 1; msk >>= 1)
      #pragma unroll
      for (int j = 0; j < 8; ++j)
        tmax[j] = fmaxf(tmax[j], __shfl_xor(tmax[j], msk, 32));

    float corr[8], psum[8];
    #pragma unroll
    for (int j = 0; j < 8; ++j) {
      const float mnew = fmaxf(mrow[j], tmax[j]);
      corr[j] = __expf(mrow[j] - mnew);
      mrow[j] = mnew;
      const float p0 = __expf(sreg[0][j] - mnew);
      const float p1 = __expf(sreg[1][j] - mnew);
      P[(half * 8 + j) * 32 + l15]      = (bf16)p0;   // ds_store
      P[(half * 8 + j) * 32 + 16 + l15] = (bf16)p1;
      psum[j] = p0 + p1;
    }
    #pragma unroll
    for (int msk = 8; msk >= 1; msk >>= 1)
      #pragma unroll
      for (int j = 0; j < 8; ++j)
        psum[j] += __shfl_xor(psum[j], msk, 32);
    #pragma unroll
    for (int j = 0; j < 8; ++j) lrow[j] = lrow[j] * corr[j] + psum[j];
    #pragma unroll
    for (int n = 0; n < 8; ++n)
      #pragma unroll
      for (int j = 0; j < 8; ++j) o[n][j] *= corr[j];

    // Re-read P in A-fragment layout (ds_load), then P(16x32) * V(32x128)
    v16bf pf = load_a_frag(P, 32, 0, 0, lane);
    #pragma unroll
    for (int n = 0; n < 8; ++n) {
      v16bf vf = load_b_frag(Vp, CS, n * 16 + l15, kb, lane);
      o[n] = wmma_bf16(pf, vf, o[n]);
    }
  }

  // epilogue: divide by row sum, store bf16 to (B,S,H*128)
  #pragma unroll
  for (int j = 0; j < 8; ++j) {
    const int s = qbase + half * 8 + j;
    const float inv = 1.0f / lrow[j];
    bf16* dst = Oc + ((size_t)(b * CS + s)) * (CH * CDV) + h * CDV;
    #pragma unroll
    for (int n = 0; n < 8; ++n)
      dst[n * 16 + l15] = (bf16)(o[n][j] * inv);
  }
}

// ---------------------------------------------------------------------------
// Host orchestration
// ---------------------------------------------------------------------------
extern "C" void kernel_launch(void* const* d_in, const int* in_sizes, int n_in,
                              void* d_out, int out_size, void* d_ws, size_t ws_size,
                              hipStream_t stream) {
  (void)in_sizes; (void)n_in; (void)out_size; (void)ws_size;

  const float* x    = (const float*)d_in[0];
  const float* fc   = (const float*)d_in[1];   // (S, 32, 2) cos/sin
  const float* wqa  = (const float*)d_in[2];   // (1536, 2048)
  const float* qnw  = (const float*)d_in[3];   // (1536,)
  const float* wqb  = (const float*)d_in[4];   // (3072, 1536)
  const float* wkva = (const float*)d_in[5];   // (576, 2048)
  const float* kvnw = (const float*)d_in[6];   // (512,)
  const float* wkvb = (const float*)d_in[7];   // (4096, 512)
  const float* wo   = (const float*)d_in[8];   // (2048, 2048)
  float* out = (float*)d_out;

  char* ws = (char*)d_ws;
  size_t off = 0;
  auto alloc = [&](size_t bytes) -> char* {
    char* p = ws + off;
    off = (off + bytes + 255) & ~(size_t)255;
    return p;
  };

  bf16*  xbf    = (bf16*)alloc((size_t)CM * CHID * 2);
  bf16*  wqab   = (bf16*)alloc((size_t)CQL * CHID * 2);
  bf16*  wqbb   = (bf16*)alloc((size_t)CH * CDQK * CQL * 2);
  bf16*  wkvab  = (bf16*)alloc((size_t)CKVP * CHID * 2);          // padded 640 rows
  bf16*  wkvbb  = (bf16*)alloc((size_t)CH * (CDN + CDV) * CKVL * 2);
  bf16*  wob    = (bf16*)alloc((size_t)CHID * CH * CDV * 2);
  float* qlat   = (float*)alloc((size_t)CM * CQL * 4);
  bf16*  qlatn  = (bf16*)alloc((size_t)CM * CQL * 2);
  float* kvall  = (float*)alloc((size_t)CM * CKVP * 4);           // padded 640 cols
  bf16*  kvlatn = (bf16*)alloc((size_t)CM * CKVL * 2);
  float* qf32   = (float*)alloc((size_t)CM * CH * CDQK * 4);
  float* kvf32  = (float*)alloc((size_t)CM * CH * (CDN + CDV) * 4);
  bf16*  Qb     = (bf16*)alloc((size_t)CB * CH * CS * CDQK * 2);
  bf16*  Kb     = (bf16*)alloc((size_t)CB * CH * CS * CDQK * 2);
  bf16*  Vt     = (bf16*)alloc((size_t)CB * CH * CDV * CS * 2 + 4096); // pad: tail reads
  bf16*  attnb  = (bf16*)alloc((size_t)CM * CH * CDV * 2);

  // 1) fp32 -> bf16 conversions (wkv_a zero-padded from 576 to 640 rows)
  cvt_bf16_pad<<<4096, 256, 0, stream>>>(x,    xbf,  (long)CM * CHID,  (long)CM * CHID);
  cvt_bf16_pad<<<2048, 256, 0, stream>>>(wqa,  wqab, (long)CQL * CHID, (long)CQL * CHID);
  cvt_bf16_pad<<<2048, 256, 0, stream>>>(wqb,  wqbb, (long)CH * CDQK * CQL,
                                                     (long)CH * CDQK * CQL);
  cvt_bf16_pad<<<1024, 256, 0, stream>>>(wkva, wkvab, (long)(CKVL + CDR) * CHID,
                                                      (long)CKVP * CHID);
  cvt_bf16_pad<<<1024, 256, 0, stream>>>(wkvb, wkvbb, (long)CH * (CDN + CDV) * CKVL,
                                                      (long)CH * (CDN + CDV) * CKVL);
  cvt_bf16_pad<<<2048, 256, 0, stream>>>(wo,   wob,  (long)CHID * CH * CDV,
                                                     (long)CHID * CH * CDV);

  // 2) low-rank projections (kv_a padded to N=640, guard-free hot loop)
  gemm_nt_bf16<<<dim3(CQL / 128, CM / 128), 256, 0, stream>>>(
      xbf, wqab, qlat, CM, CQL, CHID);
  gemm_nt_bf16<<<dim3(CKVP / 128, CM / 128), 256, 0, stream>>>(
      xbf, wkvab, kvall, CM, CKVP, CHID);

  // 3) RMSNorm + cvt
  rmsnorm_cvt<<<CM, 256, 0, stream>>>(qlat, CQL, qnw, qlatn, CQL);
  rmsnorm_cvt<<<CM, 256, 0, stream>>>(kvall, CKVP, kvnw, kvlatn, CKVL);

  // 4) up projections
  gemm_nt_bf16<<<dim3((CH * CDQK) / 128, CM / 128), 256, 0, stream>>>(
      qlatn, wqbb, qf32, CM, CH * CDQK, CQL);
  gemm_nt_bf16<<<dim3((CH * (CDN + CDV)) / 128, CM / 128), 256, 0, stream>>>(
      kvlatn, wkvbb, kvf32, CM, CH * (CDN + CDV), CKVL);

  // 5) RoPE + pack into attention layouts
  rope_q_pack<<<CM * CH, 128, 0, stream>>>(qf32, fc, Qb);
  kv_pack<<<CM * CH, 128, 0, stream>>>(kvf32, kvall, fc, Kb, Vt);

  // 6) causal flash attention (B*H*(S/16) = 4096 wave-tasks)
  mla_attn<<<(CB * CH * (CS / 16)) / ATT_WAVES, 32 * ATT_WAVES, 0, stream>>>(
      Qb, Kb, Vt, attnb);

  // 7) output projection -> f32 result
  gemm_nt_bf16<<<dim3(CHID / 128, CM / 128), 256, 0, stream>>>(
      attnb, wob, out, CM, CHID, CH * CDV);
}
